// GraphSAGELayer_48455821034228
// MI455X (gfx1250) — matrix operand order repaired and checked by reading, not compiled, not run on hardware
//
#include <hip/hip_runtime.h>
#include <hip/hip_bf16.h>

// ---------------------------------------------------------------------------
// GraphSAGE layer for MI455X (gfx1250, wave32, WMMA).
//   h = [mean_agg | x] (N x 1024, bf16)  @  [W_l | W_r]^T  (1024 x 512, bf16)
//   + b_l, then BatchNorm(node dim) -> relu -> +x
// GEMM uses v_wmma_f32_16x16x32_bf16 with a 32x64 per-wave macro-tile:
// each B fragment feeds two WMMAs (two row tiles), doubling math per load
// and halving weight re-read traffic vs a 16x64 tile.
// ---------------------------------------------------------------------------

typedef __attribute__((ext_vector_type(16))) __bf16 v16bf;
typedef __attribute__((ext_vector_type(8)))  __bf16 v8bf;
typedef __attribute__((ext_vector_type(8)))  float  v8f;

#define D_DIM   512
#define K_DIM   1024   // 2*D: [mean_agg | x] concatenated along K
#define EPS_BN  1e-5f

// ---------------------------------------------------------------------------
// 1) Per-edge scatter: agg[dst] += x[src]; cnt[dst] += 1
//    One block (128 threads) per edge, float4 per thread -> 4 atomics.
// ---------------------------------------------------------------------------
__global__ void __launch_bounds__(128)
edge_agg_kernel(const float* __restrict__ x,
                const int* __restrict__ src,
                const int* __restrict__ dst,
                float* __restrict__ agg,
                float* __restrict__ cnt)
{
    const int e = blockIdx.x;
    const int s = src[e];
    const int d = dst[e];
    const float4 xv = ((const float4*)(x + (size_t)s * D_DIM))[threadIdx.x];
    float* ag = agg + (size_t)d * D_DIM + threadIdx.x * 4;
    atomicAdd(ag + 0, xv.x);
    atomicAdd(ag + 1, xv.y);
    atomicAdd(ag + 2, xv.z);
    atomicAdd(ag + 3, xv.w);
    if (threadIdx.x == 0) atomicAdd(cnt + d, 1.0f);
}

// ---------------------------------------------------------------------------
// 2) mean + pack A = [mean_agg | x] as bf16 [N x 1024]
// ---------------------------------------------------------------------------
__global__ void __launch_bounds__(256)
pack_a_kernel(const float* __restrict__ agg,
              const float* __restrict__ cnt,
              const float* __restrict__ x,
              __bf16* __restrict__ Acat,
              int n_nodes)
{
    const size_t idx = (size_t)blockIdx.x * blockDim.x + threadIdx.x;
    const size_t total = (size_t)n_nodes * D_DIM;
    if (idx >= total) return;
    const size_t n = idx / D_DIM;
    const size_t d = idx % D_DIM;
    const float c = cnt[n];
    const float m = agg[idx] / fmaxf(c, 1.0f);
    Acat[n * K_DIM + d]         = (__bf16)m;
    Acat[n * K_DIM + D_DIM + d] = (__bf16)x[idx];
}

// ---------------------------------------------------------------------------
// 3) pack W = [W_l | W_r] as bf16, layout Wcat[col][k] (512 rows x 1024 cols)
//    B-matrix element B[k][col] = Wcat[col][k] -> contiguous-in-k per lane.
// ---------------------------------------------------------------------------
__global__ void __launch_bounds__(256)
pack_w_kernel(const float* __restrict__ W_l,
              const float* __restrict__ W_r,
              __bf16* __restrict__ Wcat)
{
    const int idx = blockIdx.x * blockDim.x + threadIdx.x;
    if (idx >= D_DIM * D_DIM) return;
    const int d = idx / D_DIM;
    const int k = idx % D_DIM;
    Wcat[(size_t)d * K_DIM + k]         = (__bf16)W_l[idx];
    Wcat[(size_t)d * K_DIM + D_DIM + k] = (__bf16)W_r[idx];
}

// ---------------------------------------------------------------------------
// 4) WMMA GEMM: h[N x 512] = Acat[N x 1024] * Wcat^T + b_l
//    Block = 256 threads = 8 waves. Block b covers rows [32b, 32b+32).
//    Wave w computes a 32-row x 64-col macro-tile: 2 row tiles x 4 col tiles
//    (columns [64w, 64w+64)), so each B fragment feeds 2 WMMAs.
//    K loop: 32 steps of 32.
//
//    Fragment layouts as loaded here (wave32):
//      A (16x32 bf16): lane 0-15 -> row = lane, K 0..15;
//                      lane 16-31 -> row = lane-16, K 16..31
//      B (32x16 bf16): col = lane&15; lanes 0-15 hold K 0..15,
//                      lanes 16-31 hold K 16..31 (contiguous in k per lane)
//      C/D (16x16 f32): VGPR r -> row r (lanes 0-15) / row r+8 (lanes 16-31),
//                       col = lane&15
// ---------------------------------------------------------------------------
__global__ void __launch_bounds__(256)
gemm_wmma_kernel(const __bf16* __restrict__ Acat,
                 const __bf16* __restrict__ Wcat,
                 const float* __restrict__ b_l,
                 float* __restrict__ h,
                 int n_nodes)
{
    const int lane  = threadIdx.x & 31;
    const int wave  = threadIdx.x >> 5;       // 0..7 = column group
    const int hiSel = lane >> 4;              // 0 or 1 (lane half)

    const int rowBlk  = blockIdx.x * 32;      // first row of this block
    const int row0    = rowBlk + (lane & 15);         // row tile 0
    const int row1r   = rowBlk + 16 + (lane & 15);    // row tile 1 (may be OOB)
    const bool tile1_valid = (rowBlk + 16) < n_nodes; // N % 16 == 0 -> uniform
    const int row1    = tile1_valid ? row1r : row0;   // clamp loads if invalid
    const int colBase = wave * 64;

    const __bf16* pA0 = Acat + (size_t)row0 * K_DIM + hiSel * 16;
    const __bf16* pA1 = Acat + (size_t)row1 * K_DIM + hiSel * 16;
    const __bf16* pB0 = Wcat + (size_t)(colBase +  0 + (lane & 15)) * K_DIM + hiSel * 16;
    const __bf16* pB1 = Wcat + (size_t)(colBase + 16 + (lane & 15)) * K_DIM + hiSel * 16;
    const __bf16* pB2 = Wcat + (size_t)(colBase + 32 + (lane & 15)) * K_DIM + hiSel * 16;
    const __bf16* pB3 = Wcat + (size_t)(colBase + 48 + (lane & 15)) * K_DIM + hiSel * 16;

    v8f acc00 = {}, acc01 = {}, acc02 = {}, acc03 = {};
    v8f acc10 = {}, acc11 = {}, acc12 = {}, acc13 = {};

    for (int k0 = 0; k0 < K_DIM; k0 += 32) {
        // A fragments: two contiguous 16B chunks of 8 bf16 each
        const v8bf a0lo = *(const v8bf*)(pA0 + k0);
        const v8bf a0hi = *(const v8bf*)(pA0 + k0 + 8);
        const v8bf a1lo = *(const v8bf*)(pA1 + k0);
        const v8bf a1hi = *(const v8bf*)(pA1 + k0 + 8);
        v16bf a0, a1;
#pragma unroll
        for (int i = 0; i < 8; ++i) {
            a0[i] = a0lo[i]; a0[8 + i] = a0hi[i];
            a1[i] = a1lo[i]; a1[8 + i] = a1hi[i];
        }

        // B fragments: one contiguous 32B chunk of 16 bf16 each;
        // each feeds both row tiles (2 WMMAs per load pair).
        const v16bf b0 = *(const v16bf*)(pB0 + k0);
        acc00 = __builtin_amdgcn_wmma_f32_16x16x32_bf16(false, a0, false, b0,
                                                        (short)0, acc00, false, false);
        acc10 = __builtin_amdgcn_wmma_f32_16x16x32_bf16(false, a1, false, b0,
                                                        (short)0, acc10, false, false);
        const v16bf b1 = *(const v16bf*)(pB1 + k0);
        acc01 = __builtin_amdgcn_wmma_f32_16x16x32_bf16(false, a0, false, b1,
                                                        (short)0, acc01, false, false);
        acc11 = __builtin_amdgcn_wmma_f32_16x16x32_bf16(false, a1, false, b1,
                                                        (short)0, acc11, false, false);
        const v16bf b2 = *(const v16bf*)(pB2 + k0);
        acc02 = __builtin_amdgcn_wmma_f32_16x16x32_bf16(false, a0, false, b2,
                                                        (short)0, acc02, false, false);
        acc12 = __builtin_amdgcn_wmma_f32_16x16x32_bf16(false, a1, false, b2,
                                                        (short)0, acc12, false, false);
        const v16bf b3 = *(const v16bf*)(pB3 + k0);
        acc03 = __builtin_amdgcn_wmma_f32_16x16x32_bf16(false, a0, false, b3,
                                                        (short)0, acc03, false, false);
        acc13 = __builtin_amdgcn_wmma_f32_16x16x32_bf16(false, a1, false, b3,
                                                        (short)0, acc13, false, false);
    }

    // Epilogue: bias fused. D layout -> row = rbase + r, col = colBase + lane&15
    const int c0  = colBase + (lane & 15);
    const float bl0 = b_l[c0];
    const float bl1 = b_l[c0 + 16];
    const float bl2 = b_l[c0 + 32];
    const float bl3 = b_l[c0 + 48];

    {
        const int rbase = rowBlk + hiSel * 8;
#pragma unroll
        for (int r = 0; r < 8; ++r) {
            const size_t rr = (size_t)(rbase + r) * D_DIM;
            h[rr + c0]      = acc00[r] + bl0;
            h[rr + c0 + 16] = acc01[r] + bl1;
            h[rr + c0 + 32] = acc02[r] + bl2;
            h[rr + c0 + 48] = acc03[r] + bl3;
        }
    }
    if (tile1_valid) {
        const int rbase = rowBlk + 16 + hiSel * 8;
#pragma unroll
        for (int r = 0; r < 8; ++r) {
            const size_t rr = (size_t)(rbase + r) * D_DIM;
            h[rr + c0]      = acc10[r] + bl0;
            h[rr + c0 + 16] = acc11[r] + bl1;
            h[rr + c0 + 32] = acc12[r] + bl2;
            h[rr + c0 + 48] = acc13[r] + bl3;
        }
    }
}

// ---------------------------------------------------------------------------
// 5) BatchNorm stats: per-column sum & sumsq over N rows (thread = column)
// ---------------------------------------------------------------------------
__global__ void __launch_bounds__(512)
bn_stats_kernel(const float* __restrict__ h,
                float* __restrict__ colsum,
                float* __restrict__ colsq,
                int n_nodes)
{
    const int col = threadIdx.x;   // 0..511
    float s = 0.0f, s2 = 0.0f;
    for (int r = blockIdx.x; r < n_nodes; r += gridDim.x) {
        const float v = h[(size_t)r * D_DIM + col];
        s  += v;
        s2 += v * v;
    }
    atomicAdd(colsum + col, s);
    atomicAdd(colsq + col, s2);
}

__global__ void __launch_bounds__(512)
bn_finalize_kernel(const float* __restrict__ colsum,
                   const float* __restrict__ colsq,
                   float* __restrict__ mu,
                   float* __restrict__ rsig,
                   int n_nodes)
{
    const int col = threadIdx.x;
    const float inv_n = 1.0f / (float)n_nodes;
    const float m   = colsum[col] * inv_n;
    const float var = colsq[col] * inv_n - m * m;
    mu[col]   = m;
    rsig[col] = rsqrtf(var + EPS_BN);
}

// ---------------------------------------------------------------------------
// 6) normalize + relu + residual
// ---------------------------------------------------------------------------
__global__ void __launch_bounds__(256)
bn_apply_kernel(const float* __restrict__ h,
                const float* __restrict__ x,
                const float* __restrict__ mu,
                const float* __restrict__ rsig,
                const float* __restrict__ gamma,
                const float* __restrict__ beta,
                float* __restrict__ out,
                int n_nodes)
{
    const size_t idx = (size_t)blockIdx.x * blockDim.x + threadIdx.x;
    const size_t total = (size_t)n_nodes * D_DIM;
    if (idx >= total) return;
    const int col = (int)(idx % D_DIM);
    const float v = (h[idx] - mu[col]) * rsig[col] * gamma[col] + beta[col];
    out[idx] = fmaxf(v, 0.0f) + x[idx];
}

// ---------------------------------------------------------------------------
// Launch
// ---------------------------------------------------------------------------
static inline size_t align256(size_t v) { return (v + 255) & ~(size_t)255; }

extern "C" void kernel_launch(void* const* d_in, const int* in_sizes, int n_in,
                              void* d_out, int out_size, void* d_ws, size_t ws_size,
                              hipStream_t stream) {
    const float* x          = (const float*)d_in[0];
    const int*   edge_index = (const int*)  d_in[1];
    const float* W_l        = (const float*)d_in[2];
    const float* b_l        = (const float*)d_in[3];
    const float* W_r        = (const float*)d_in[4];
    const float* gamma      = (const float*)d_in[5];
    const float* beta       = (const float*)d_in[6];
    float*       out        = (float*)d_out;

    const int N = in_sizes[0] / D_DIM;     // 50000
    const int E = in_sizes[1] / 2;         // 400000
    const int* src = edge_index;
    const int* dst = edge_index + E;

    // ---- workspace carve-up (agg and h share one region) ----
    char* ws = (char*)d_ws;
    size_t off = 0;
    float*  agg_h = (float*)(ws + off);  off += align256((size_t)N * D_DIM * sizeof(float));
    float*  cnt   = (float*)(ws + off);  off += align256((size_t)N * sizeof(float));
    __bf16* Acat  = (__bf16*)(ws + off); off += align256((size_t)N * K_DIM * sizeof(__bf16));
    __bf16* Wcat  = (__bf16*)(ws + off); off += align256((size_t)D_DIM * K_DIM * sizeof(__bf16));
    float*  csum  = (float*)(ws + off);  off += align256((size_t)D_DIM * sizeof(float));
    float*  csq   = (float*)(ws + off);  off += align256((size_t)D_DIM * sizeof(float));
    float*  mu    = (float*)(ws + off);  off += align256((size_t)D_DIM * sizeof(float));
    float*  rsig  = (float*)(ws + off);  off += align256((size_t)D_DIM * sizeof(float));
    (void)ws_size;

    // zero the accumulators (graph-capturable)
    hipMemsetAsync(agg_h, 0, (size_t)N * D_DIM * sizeof(float), stream);
    hipMemsetAsync(cnt,   0, (size_t)N * sizeof(float), stream);
    hipMemsetAsync(csum,  0, (size_t)D_DIM * sizeof(float), stream);
    hipMemsetAsync(csq,   0, (size_t)D_DIM * sizeof(float), stream);

    // 1) edge scatter-mean accumulation
    edge_agg_kernel<<<E, 128, 0, stream>>>(x, src, dst, agg_h, cnt);

    // 2) pack A = [mean | x] bf16
    {
        const size_t total = (size_t)N * D_DIM;
        pack_a_kernel<<<(unsigned)((total + 255) / 256), 256, 0, stream>>>(
            agg_h, cnt, x, Acat, N);
    }

    // 3) pack W = [W_l | W_r] bf16
    pack_w_kernel<<<(D_DIM * D_DIM + 255) / 256, 256, 0, stream>>>(W_l, W_r, Wcat);

    // 4) GEMM (writes h into the agg region; agg no longer needed)
    {
        const int rowBlocks = (N + 31) / 32;    // 1563 (last block: 1 valid tile)
        gemm_wmma_kernel<<<rowBlocks, 256, 0, stream>>>(Acat, Wcat, b_l, agg_h, N);
    }

    // 5) BN stats + finalize
    bn_stats_kernel<<<256, 512, 0, stream>>>(agg_h, csum, csq, N);
    bn_finalize_kernel<<<1, 512, 0, stream>>>(csum, csq, mu, rsig, N);

    // 6) normalize + relu + residual
    {
        const size_t total = (size_t)N * D_DIM;
        bn_apply_kernel<<<(unsigned)((total + 255) / 256), 256, 0, stream>>>(
            agg_h, x, mu, rsig, gamma, beta, out, N);
    }
}
